// MultiHeadedSelfAttention_5179730559275
// MI455X (gfx1250) — matrix operand order repaired and checked by reading, not compiled, not run on hardware
//
#include <hip/hip_runtime.h>
#include <hip/hip_bf16.h>
#include <math.h>

typedef __attribute__((ext_vector_type(16))) __bf16 v16bf;
typedef __attribute__((ext_vector_type(2)))  __bf16 v2bf;
typedef __attribute__((ext_vector_type(8)))  float  v8f;
typedef __attribute__((ext_vector_type(2)))  float  v2f;
typedef __attribute__((ext_vector_type(4)))  int    v4i;

#define NB   16
#define NC   256
#define NO   256
#define NHW  4096   // 64*64

// 32-byte WMMA 16-bit fragment viewed as two 128-bit LDS loads
union Frag16 { v16bf v; uint4 q[2]; };

// packed f32x2 -> bf16x2 (single v_cvt_pk_bf16_f32)
__device__ __forceinline__ unsigned pack_bf16(float a, float b) {
    v2f f; f.x = a; f.y = b;
    union { v2bf v; unsigned u; } p;
    p.v = __builtin_convertvector(f, v2bf);
    return p.u;
}

__device__ __forceinline__ void lds_fence_wave() {
    asm volatile("s_wait_dscnt 0x0" ::: "memory");
}

__device__ __forceinline__ void async_wait0() {
#if __has_builtin(__builtin_amdgcn_s_wait_asynccnt)
    __builtin_amdgcn_s_wait_asynccnt(0);
#else
    asm volatile("s_wait_asynccnt 0x0" ::: "memory");
#endif
}

// async 16B global -> LDS copy (ASYNCcnt path); falls back to a VGPR copy
__device__ __forceinline__ void async_copy_b128(const void* g, void* l) {
#if __has_builtin(__builtin_amdgcn_global_load_async_to_lds_b128)
    __builtin_amdgcn_global_load_async_to_lds_b128(
        (__attribute__((address_space(1))) v4i*)g,
        (__attribute__((address_space(3))) v4i*)l, 0, 0);
#else
    *(uint4*)l = *(const uint4*)g;
#endif
}

// ---------------------------------------------------------------------------
// Kernel 1: fused QKV 1x1-conv.  X_b is [C, HW] row-major.
// Q_b = Wq @ X_b + bq -> bf16 [B, O, HW]; same for K, V.
// 256 threads = 8 waves; block tile 64(O) x 64(HW); each wave owns TWO
// 16x16 tiles per matrix (6 accumulators) so the staged X chunk is reused
// by 6 WMMAs and only 4 m-blocks re-read X from L2.
// LDS layouts are fragment-friendly: A = [m][k] row-major, B = [n][k].
// ---------------------------------------------------------------------------
__global__ __launch_bounds__(256)
void qkv_proj_kernel(const float* __restrict__ x,
                     const float* __restrict__ wq, const float* __restrict__ bq,
                     const float* __restrict__ wk, const float* __restrict__ bk,
                     const float* __restrict__ wv, const float* __restrict__ bv,
                     __bf16* __restrict__ outq, __bf16* __restrict__ outk,
                     __bf16* __restrict__ outv)
{
    __shared__ __attribute__((aligned(16))) __bf16 sXt[64 * 40];   // [n][k], stride 40 (80B, 16-aligned)
    __shared__ __attribute__((aligned(16))) __bf16 sW[3][64 * 32]; // [m][k]

    const int n0   = blockIdx.x * 64;
    const int m0   = blockIdx.y * 64;
    const int b    = blockIdx.z;
    const int tid  = threadIdx.x;
    const int lane = tid & 31;
    const int wid  = tid >> 5;
    const int tmw  = (wid >> 2) * 32;     // wave's M pair: tmw and tmw+16
    const int tn   = (wid & 3) * 16;      // 0,16,32,48
    const int half = lane >> 4;
    const int l15  = lane & 15;

    const float* xb = x + (size_t)b * NC * NHW;

    v8f accq[2] = {}; v8f acck[2] = {}; v8f accv[2] = {};

    for (int kc = 0; kc < NC; kc += 32) {
        // stage X chunk [k=32][n=64] transposed -> sXt[n][k].
        // 1024 k-adjacent pairs; coalesced b32 loads, packed b32 LDS stores.
        #pragma unroll
        for (int j = 0; j < 4; ++j) {
            int p = tid + j * 256;
            int c = p & 63, k = (p >> 6) * 2;
            float e0 = xb[(size_t)(kc + k)     * NHW + n0 + c];
            float e1 = xb[(size_t)(kc + k + 1) * NHW + n0 + c];
            *(unsigned*)&sXt[c * 40 + k] = pack_bf16(e0, e1);
        }
        // stage three W chunks [m=64][k=32]; float2 loads (coalesced b64),
        // packed b32 LDS stores.
        #pragma unroll
        for (int j = 0; j < 4; ++j) {
            int p = tid + j * 256;
            int k = (p & 15) * 2, m = p >> 4;
            size_t g = (size_t)(m0 + m) * NC + kc + k;
            float2 q2 = *(const float2*)&wq[g];
            float2 k2 = *(const float2*)&wk[g];
            float2 v2 = *(const float2*)&wv[g];
            *(unsigned*)&sW[0][m * 32 + k] = pack_bf16(q2.x, q2.y);
            *(unsigned*)&sW[1][m * 32 + k] = pack_bf16(k2.x, k2.y);
            *(unsigned*)&sW[2][m * 32 + k] = pack_bf16(v2.x, v2.y);
        }
        __syncthreads();

        // B fragment: k = 8h..8h+7 and 16+8h..16+8h+7 per lane -> 2 x ds_load_b128
        Frag16 fb;
        {
            const uint4* p = (const uint4*)&sXt[(tn + l15) * 40 + half * 8];
            fb.q[0] = p[0]; fb.q[1] = p[2];
        }
        // A fragments for both m sub-tiles of all three matrices
        #pragma unroll
        for (int s = 0; s < 2; ++s) {
            int m = l15 + tmw + s * 16, koff = half * 8;
            Frag16 faq, fak, fav;
            const uint4* p0 = (const uint4*)&sW[0][m * 32 + koff];
            const uint4* p1 = (const uint4*)&sW[1][m * 32 + koff];
            const uint4* p2 = (const uint4*)&sW[2][m * 32 + koff];
            faq.q[0] = p0[0]; faq.q[1] = p0[2];
            fak.q[0] = p1[0]; fak.q[1] = p1[2];
            fav.q[0] = p2[0]; fav.q[1] = p2[2];
            accq[s] = __builtin_amdgcn_wmma_f32_16x16x32_bf16(false, faq.v, false, fb.v, (short)0, accq[s], false, false);
            acck[s] = __builtin_amdgcn_wmma_f32_16x16x32_bf16(false, fak.v, false, fb.v, (short)0, acck[s], false, false);
            accv[s] = __builtin_amdgcn_wmma_f32_16x16x32_bf16(false, fav.v, false, fb.v, (short)0, accv[s], false, false);
        }
        __syncthreads();
    }

    // epilogue: bias rows are contiguous per sub-tile -> float4 loads
    const int ncol = n0 + tn + l15;
    #pragma unroll
    for (int s = 0; s < 2; ++s) {
        const int o0 = m0 + tmw + s * 16 + half * 8;
        const float4 q0 = *(const float4*)&bq[o0], q1 = *(const float4*)&bq[o0 + 4];
        const float4 k0 = *(const float4*)&bk[o0], k1 = *(const float4*)&bk[o0 + 4];
        const float4 v0 = *(const float4*)&bv[o0], v1 = *(const float4*)&bv[o0 + 4];
        const float bqr[8] = {q0.x, q0.y, q0.z, q0.w, q1.x, q1.y, q1.z, q1.w};
        const float bkr[8] = {k0.x, k0.y, k0.z, k0.w, k1.x, k1.y, k1.z, k1.w};
        const float bvr[8] = {v0.x, v0.y, v0.z, v0.w, v1.x, v1.y, v1.z, v1.w};
        #pragma unroll
        for (int r = 0; r < 8; ++r) {
            size_t dst = ((size_t)b * NO + o0 + r) * NHW + ncol;
            outq[dst] = (__bf16)(accq[s][r] + bqr[r]);
            outk[dst] = (__bf16)(acck[s][r] + bkr[r]);
            outv[dst] = (__bf16)(accv[s][r] + bvr[r]);
        }
    }
}

// ---------------------------------------------------------------------------
// Kernel 2: one workgroup (4 waves) per (batch, channel) 64x64 plane.
// S = Q K^T / sqrt(32); P = softmax_rows(S) (in registers); out = P V.
// ---------------------------------------------------------------------------
__global__ __launch_bounds__(128)
void attn_kernel(const __bf16* __restrict__ Q, const __bf16* __restrict__ K,
                 const __bf16* __restrict__ V, float* __restrict__ out)
{
    __shared__ __attribute__((aligned(16))) __bf16 sQ[64 * 64];      // [h][w]
    __shared__ __attribute__((aligned(16))) __bf16 sK[64 * 64];      // [g][w] == B[n][k]
    __shared__ __attribute__((aligned(16))) __bf16 sVt[64 * 72];     // [w][g] padded (144B rows)
    __shared__ __attribute__((aligned(16))) __bf16 sP[4][16 * 64];   // per-wave P stripe [m][g]

    const int tid  = threadIdx.x;
    const int lane = tid & 31;
    const int w    = tid >> 5;
    const int half = lane >> 4;
    const int l15  = lane & 15;
    const size_t plane = (size_t)blockIdx.x * NHW;

    // ---- staging ------------------------------------------------------------
    {
        const uint4* gq = (const uint4*)(Q + plane);
        const uint4* gk = (const uint4*)(K + plane);
        const uint4* gv = (const uint4*)(V + plane);
        uint4* q4 = (uint4*)sQ;
        uint4* k4 = (uint4*)sK;
        #pragma unroll
        for (int j = 0; j < 4; ++j) {           // 512 x b128 over 128 threads
            int i = tid + j * 128;
            async_copy_b128(gq + i, q4 + i);    // ASYNCcnt path: no VGPR round-trip
            async_copy_b128(gk + i, k4 + i);
        }
        #pragma unroll
        for (int j = 0; j < 4; ++j) {           // V transpose: [g][w] -> sVt[w][g]
            int i = tid + j * 128;
            union { uint4 q; __bf16 h[8]; } u;
            u.q = gv[i];
            int g  = i >> 3;                    // row of V plane
            int w0 = (i & 7) * 8;               // starting column
            #pragma unroll
            for (int jj = 0; jj < 8; ++jj)
                sVt[(w0 + jj) * 72 + g] = u.h[jj];
        }
    }
    async_wait0();
    __syncthreads();

    const int m0 = w * 16;              // this wave's score-row stripe

    // ---- scores: S[h,g] = sum_w Q[h,w] K[g,w] -------------------------------
    v8f acc[4] = {};
    #pragma unroll
    for (int kc = 0; kc < 64; kc += 32) {
        Frag16 fa;                      // Q stripe 16x32, A layout
        {
            const uint4* p = (const uint4*)&sQ[(m0 + l15) * 64 + half * 8 + kc];
            fa.q[0] = p[0]; fa.q[1] = p[2];
        }
        #pragma unroll
        for (int t = 0; t < 4; ++t) {   // g-tiles: B[k=w, n=g] = K[g, w]
            Frag16 fb;
            const uint4* p = (const uint4*)&sK[(t * 16 + l15) * 64 + half * 8 + kc];
            fb.q[0] = p[0]; fb.q[1] = p[2];
            acc[t] = __builtin_amdgcn_wmma_f32_16x16x32_bf16(false, fa.v, false, fb.v, (short)0, acc[t], false, false);
        }
    }

    // ---- softmax fully in registers ----------------------------------------
    // C/D layout: acc[t][r] = S[row = r + half*8][col = l15 + 16t]; each row
    // lives in the 16 lanes of one half-wave -> shfl_xor 1/2/4/8 reduces it.
    {
        const float scale = 0.1767766952966369f;   // 1/sqrt(32)
        #pragma unroll
        for (int r = 0; r < 8; ++r) {
            float mx = fmaxf(fmaxf(acc[0][r], acc[1][r]), fmaxf(acc[2][r], acc[3][r]));
            mx = fmaxf(mx, __shfl_xor(mx, 1, 32));
            mx = fmaxf(mx, __shfl_xor(mx, 2, 32));
            mx = fmaxf(mx, __shfl_xor(mx, 4, 32));
            mx = fmaxf(mx, __shfl_xor(mx, 8, 32));
            float e0 = __expf((acc[0][r] - mx) * scale);
            float e1 = __expf((acc[1][r] - mx) * scale);
            float e2 = __expf((acc[2][r] - mx) * scale);
            float e3 = __expf((acc[3][r] - mx) * scale);
            float sum = (e0 + e1) + (e2 + e3);
            sum += __shfl_xor(sum, 1, 32);
            sum += __shfl_xor(sum, 2, 32);
            sum += __shfl_xor(sum, 4, 32);
            sum += __shfl_xor(sum, 8, 32);
            float inv = 1.0f / sum;
            int row = r + half * 8;
            sP[w][row * 64 + 0  + l15] = (__bf16)(e0 * inv);
            sP[w][row * 64 + 16 + l15] = (__bf16)(e1 * inv);
            sP[w][row * 64 + 32 + l15] = (__bf16)(e2 * inv);
            sP[w][row * 64 + 48 + l15] = (__bf16)(e3 * inv);
        }
    }
    lds_fence_wave();   // same-wave LDS RAW (P written above, read as A below)

    // ---- ctx = P V (stripe 16 x 64) ----------------------------------------
    v8f acc2[4] = {};
    #pragma unroll
    for (int kc = 0; kc < 64; kc += 32) {
        Frag16 fa;                      // P stripe 16x32, A layout
        {
            const uint4* p = (const uint4*)&sP[w][l15 * 64 + half * 8 + kc];
            fa.q[0] = p[0]; fa.q[1] = p[2];
        }
        #pragma unroll
        for (int t = 0; t < 4; ++t) {   // w-tiles: B[k=g, n=w] = V[g, w] = sVt[w][g]
            Frag16 fb;
            const uint4* p = (const uint4*)&sVt[(t * 16 + l15) * 72 + half * 8 + kc];
            fb.q[0] = p[0]; fb.q[1] = p[2];
            acc2[t] = __builtin_amdgcn_wmma_f32_16x16x32_bf16(false, fa.v, false, fb.v, (short)0, acc2[t], false, false);
        }
    }

    #pragma unroll
    for (int t = 0; t < 4; ++t)
        #pragma unroll
        for (int r = 0; r < 8; ++r)
            out[plane + (size_t)(m0 + r + half * 8) * 64 + t * 16 + l15] = acc2[t][r];
}

// ---------------------------------------------------------------------------
extern "C" void kernel_launch(void* const* d_in, const int* in_sizes, int n_in,
                              void* d_out, int out_size, void* d_ws, size_t ws_size,
                              hipStream_t stream)
{
    const float* x  = (const float*)d_in[0];
    const float* wq = (const float*)d_in[1];
    const float* bq = (const float*)d_in[2];
    const float* wk = (const float*)d_in[3];
    const float* bk = (const float*)d_in[4];
    const float* wv = (const float*)d_in[5];
    const float* bv = (const float*)d_in[6];
    float* out = (float*)d_out;

    const size_t planeElems = (size_t)NB * NO * NHW;   // 16.78M bf16 per tensor
    __bf16* wsQ = (__bf16*)d_ws;
    __bf16* wsK = wsQ + planeElems;
    __bf16* wsV = wsK + planeElems;

    dim3 g1(NHW / 64, NO / 64, NB);       // (64, 4, 16)
    qkv_proj_kernel<<<g1, 256, 0, stream>>>(x, wq, bq, wk, bk, wv, bv, wsQ, wsK, wsV);

    attn_kernel<<<NB * NO, 128, 0, stream>>>(wsQ, wsK, wsV, out);
}
